// Naive_LSTM_14474039787794
// MI455X (gfx1250) — matrix operand order repaired and checked by reading, not compiled, not run on hardware
//
#include <hip/hip_runtime.h>

typedef __attribute__((ext_vector_type(16))) _Float16 v16h;
typedef __attribute__((ext_vector_type(8)))  _Float16 v8h;
typedef __attribute__((ext_vector_type(8)))  float    v8f;

#define T_LEN 1024
#define BATCH 2048
#define H1C   51
#define NG1   204      // 4*H1
#define NG1P  208      // padded to 13 tiles of 16
#define KP    64       // K padded 51 -> 64 (two 16x16x32 WMMA K-chunks)
#define NT1   13

__device__ __forceinline__ float sig_(float x)  { return 1.f / (1.f + __expf(-x)); }
__device__ __forceinline__ float tanh_(float x) { return 1.f - 2.f / (__expf(2.f * x) + 1.f); }

extern "C" __global__ __launch_bounds__(32, 1)
void lstm2_wmma_kernel(const float* __restrict__ input,
                       const float* __restrict__ W_ih1,
                       const float* __restrict__ W_hh1,
                       const float* __restrict__ b_ih1,
                       const float* __restrict__ b_hh1,
                       const float* __restrict__ W_ih2,
                       const float* __restrict__ W_hh2,
                       const float* __restrict__ b_ih2,
                       const float* __restrict__ b_hh2,
                       float* __restrict__ out)
{
    // --- persistent on-chip state for 16 batch rows (one wave) ---
    __shared__ __align__(32) _Float16 sWB [NG1P * KP]; // W_hh1 as f16 B-tiles [n][k]
    __shared__ __align__(32) _Float16 sWB2[16 * KP];   // W_ih2  as f16 B-tile  [gate][k]
    __shared__ __align__(32) _Float16 sH1 [16 * KP];   // h1 (f16, A-frag source)
    __shared__ __align__(32) _Float16 sC1h[16 * KP];   // c1 (f16, A-frag source for layer 2)
    __shared__ float sC1[16 * 52];                     // c1 state, f32
    __shared__ float sG1[16 * NG1P];                   // layer-1 gates scratch
    __shared__ float sG2[16 * 16];                     // layer-2 gates scratch
    __shared__ float sB1[NG1P];
    __shared__ float sWi1[NG1P];
    __shared__ float sX[16];

    const int lane = threadIdx.x;
    const int half = lane >> 4;     // K-half select per WMMA A/B layout
    const int colL = lane & 15;     // row (A) / column (B,C,D) index
    const int m0   = blockIdx.x * 16;

    // ---- one-time init: convert + pad weights into LDS, zero state ----
    for (int i = lane; i < NG1P * KP; i += 32) {
        int n = i >> 6, k = i & 63;
        sWB[i] = (n < NG1 && k < H1C) ? (_Float16)W_hh1[n * H1C + k] : (_Float16)0.f;
    }
    for (int i = lane; i < 16 * KP; i += 32) {
        int n = i >> 6, k = i & 63;
        sWB2[i] = (n < 4 && k < H1C) ? (_Float16)W_ih2[n * H1C + k] : (_Float16)0.f;
        sH1[i]  = (_Float16)0.f;
        sC1h[i] = (_Float16)0.f;
    }
    for (int i = lane; i < NG1P; i += 32) {
        sB1[i]  = (i < NG1) ? (b_ih1[i] + b_hh1[i]) : 0.f;
        sWi1[i] = (i < NG1) ? W_ih1[i] : 0.f;
    }
    for (int i = lane; i < 16 * 52; i += 32) sC1[i] = 0.f;

    float b2g[4], whh2r[4];
#pragma unroll
    for (int g = 0; g < 4; ++g) { b2g[g] = b_ih2[g] + b_hh2[g]; whh2r[g] = W_hh2[g]; }
    float h2 = 0.f, c2 = 0.f;      // layer-2 state, owned by lanes 0..15

    __syncthreads();

    for (int t = 0; t < T_LEN; ++t) {
        if (lane < 16) sX[lane] = input[(size_t)(m0 + lane) * T_LEN + t];
        __syncthreads();  // S1: sX ready; sH1 from previous step ready

        float xr[8];
#pragma unroll
        for (int v = 0; v < 8; ++v) xr[v] = sX[half * 8 + v];

        // A-fragments of h1 (16x64 f16): lane<16 holds K 0-7/16-23, lane>=16 K 8-15/24-31
        v16h a0, a1;
        {
            union { v16h v; v8h h[2]; } u0, u1;
            u0.h[0] = *(const v8h*)&sH1[colL * KP +      half * 8];
            u0.h[1] = *(const v8h*)&sH1[colL * KP + 16 + half * 8];
            u1.h[0] = *(const v8h*)&sH1[colL * KP + 32 + half * 8];
            u1.h[1] = *(const v8h*)&sH1[colL * KP + 48 + half * 8];
            a0 = u0.v; a1 = u1.v;
        }

        // gates1[16 x 208] = h1 @ W_hh1^T  (13 N-tiles x 2 K-chunks of WMMA)
#pragma unroll
        for (int n = 0; n < NT1; ++n) {
            const int col = n * 16 + colL;
            union { v16h v; v8h h[2]; } ub0, ub1;
            ub0.h[0] = *(const v8h*)&sWB[col * KP +      half * 8];
            ub0.h[1] = *(const v8h*)&sWB[col * KP + 16 + half * 8];
            ub1.h[0] = *(const v8h*)&sWB[col * KP + 32 + half * 8];
            ub1.h[1] = *(const v8h*)&sWB[col * KP + 48 + half * 8];
            v16h b0 = ub0.v;
            v16h b1 = ub1.v;
            v8f acc = {};
            acc = __builtin_amdgcn_wmma_f32_16x16x32_f16(false, a0, false, b0,
                                                         (short)0, acc, false, false);
            acc = __builtin_amdgcn_wmma_f32_16x16x32_f16(false, a1, false, b1,
                                                         (short)0, acc, false, false);
            const float bias = sB1[col];
            const float wi   = sWi1[col];
#pragma unroll
            for (int v = 0; v < 8; ++v)
                sG1[(half * 8 + v) * NG1P + col] = acc[v] + bias + xr[v] * wi;
        }
        __syncthreads();  // S2: gates1 ready

        // layer-1 cell update: row = colL, units u = half, half+2, ...
#pragma unroll 1
        for (int u = half; u < H1C; u += 2) {
            const int r = colL;
            float gi = sG1[r * NG1P + u];
            float gf = sG1[r * NG1P + H1C + u];
            float gg = sG1[r * NG1P + 2 * H1C + u];
            float go = sG1[r * NG1P + 3 * H1C + u];
            float cn = sig_(gf) * sC1[r * 52 + u] + sig_(gi) * tanh_(gg);
            float hn = sig_(go) * tanh_(cn);
            sC1[r * 52 + u]  = cn;
            sH1[r * KP + u]  = (_Float16)hn;
            sC1h[r * KP + u] = (_Float16)cn;   // layer-2 input is c1 (reference quirk)
        }
        __syncthreads();  // S3: c1 (f16) ready

        // layer 2: gates2[16 x 4] = c1 @ W_ih2^T  (one tile, 2 WMMAs)
        {
            union { v16h v; v8h h[2]; } u0, u1;
            u0.h[0] = *(const v8h*)&sC1h[colL * KP +      half * 8];
            u0.h[1] = *(const v8h*)&sC1h[colL * KP + 16 + half * 8];
            u1.h[0] = *(const v8h*)&sC1h[colL * KP + 32 + half * 8];
            u1.h[1] = *(const v8h*)&sC1h[colL * KP + 48 + half * 8];
            union { v16h v; v8h h[2]; } ub0, ub1;
            ub0.h[0] = *(const v8h*)&sWB2[colL * KP +      half * 8];
            ub0.h[1] = *(const v8h*)&sWB2[colL * KP + 16 + half * 8];
            ub1.h[0] = *(const v8h*)&sWB2[colL * KP + 32 + half * 8];
            ub1.h[1] = *(const v8h*)&sWB2[colL * KP + 48 + half * 8];
            v16h b0 = ub0.v;
            v16h b1 = ub1.v;
            v8f acc = {};
            acc = __builtin_amdgcn_wmma_f32_16x16x32_f16(false, u0.v, false, b0,
                                                         (short)0, acc, false, false);
            acc = __builtin_amdgcn_wmma_f32_16x16x32_f16(false, u1.v, false, b1,
                                                         (short)0, acc, false, false);
            if (colL < 4) {
#pragma unroll
                for (int v = 0; v < 8; ++v)
                    sG2[(half * 8 + v) * 16 + colL] = acc[v];
            }
        }
        __syncthreads();  // S4: gates2 ready

        if (lane < 16) {
            const int r = lane;
            float gi = sG2[r * 16 + 0] + b2g[0] + whh2r[0] * h2;
            float gf = sG2[r * 16 + 1] + b2g[1] + whh2r[1] * h2;
            float gg = sG2[r * 16 + 2] + b2g[2] + whh2r[2] * h2;
            float go = sG2[r * 16 + 3] + b2g[3] + whh2r[3] * h2;
            float cn = sig_(gf) * c2 + sig_(gi) * tanh_(gg);
            h2 = sig_(go) * tanh_(cn);
            c2 = cn;
            out[(size_t)(m0 + r) * T_LEN + t] = cn;  // output is layer-2 cell state
        }
        // next S1 re-separates sX writes from this step's (already done) reads
    }
}

extern "C" void kernel_launch(void* const* d_in, const int* in_sizes, int n_in,
                              void* d_out, int out_size, void* d_ws, size_t ws_size,
                              hipStream_t stream) {
    (void)in_sizes; (void)n_in; (void)out_size; (void)d_ws; (void)ws_size;
    const float* input = (const float*)d_in[0];
    const float* W_ih1 = (const float*)d_in[1];
    const float* W_hh1 = (const float*)d_in[2];
    const float* b_ih1 = (const float*)d_in[3];
    const float* b_hh1 = (const float*)d_in[4];
    const float* W_ih2 = (const float*)d_in[5];
    const float* W_hh2 = (const float*)d_in[6];
    const float* b_ih2 = (const float*)d_in[7];
    const float* b_hh2 = (const float*)d_in[8];
    float* out = (float*)d_out;

    lstm2_wmma_kernel<<<BATCH / 16, 32, 0, stream>>>(
        input, W_ih1, W_hh1, b_ih1, b_hh1, W_ih2, W_hh2, b_ih2, b_hh2, out);
}
